// myAttention_23115513987248
// MI455X (gfx1250) — matrix operand order repaired and checked by reading, not compiled, not run on hardware
//
#include <hip/hip_runtime.h>
#include <hip/hip_bf16.h>

typedef __attribute__((ext_vector_type(16))) _Float16 v16h;
typedef __attribute__((ext_vector_type(8)))  _Float16 v8h;
typedef __attribute__((ext_vector_type(8)))  float    v8f;

static __device__ __forceinline__ v16h cat_v8h(v8h lo, v8h hi) {
  v16h r;
#pragma unroll
  for (int i = 0; i < 8; ++i) { r[i] = lo[i]; r[i + 8] = hi[i]; }
  return r;
}

// ---------------------------------------------------------------------------
// Tiled transpose: src [rows, cols] -> dst [cols, rows], per batch (blockIdx.z)
// ---------------------------------------------------------------------------
__global__ void transpose_kernel(const float* __restrict__ src, float* __restrict__ dst,
                                 int rows, int cols) {
  __shared__ float tile[32][33];
  const size_t bo = (size_t)blockIdx.z * rows * cols;
  const float* s = src + bo;
  float* d = dst + bo;
  const int c0 = blockIdx.x * 32, r0 = blockIdx.y * 32;
  for (int rr = threadIdx.y; rr < 32; rr += 8)
    tile[rr][threadIdx.x] = s[(size_t)(r0 + rr) * cols + c0 + threadIdx.x];
  __syncthreads();
  for (int rr = threadIdx.y; rr < 32; rr += 8)
    d[(size_t)(c0 + rr) * rows + r0 + threadIdx.x] = tile[threadIdx.x][rr];
}

// ---------------------------------------------------------------------------
// f32 -> f16 elementwise (weight conversion)
// ---------------------------------------------------------------------------
__global__ void f32_to_f16(const float* __restrict__ s, _Float16* __restrict__ d, int n) {
  int i = blockIdx.x * blockDim.x + threadIdx.x;
  if (i < n) d[i] = (_Float16)s[i];
}

// ---------------------------------------------------------------------------
// LayerNorm over C=512, one wave per token, f16 output
// ---------------------------------------------------------------------------
__global__ __launch_bounds__(256) void layernorm_kernel(const float* __restrict__ X,
                                                        const float* __restrict__ wgt,
                                                        const float* __restrict__ bia,
                                                        _Float16* __restrict__ Y) {
  const int lane = threadIdx.x & 31;
  const int wv = threadIdx.x >> 5;
  const int token = blockIdx.x * 8 + wv;
  const float* row = X + (size_t)token * 512;
  float xv[16];
  float s1 = 0.f, s2 = 0.f;
#pragma unroll
  for (int i = 0; i < 4; ++i) {
    float4 t = ((const float4*)row)[lane + i * 32];
    xv[4*i+0] = t.x; xv[4*i+1] = t.y; xv[4*i+2] = t.z; xv[4*i+3] = t.w;
    s1 += t.x + t.y + t.z + t.w;
    s2 += t.x*t.x + t.y*t.y + t.z*t.z + t.w*t.w;
  }
#pragma unroll
  for (int off = 16; off; off >>= 1) { s1 += __shfl_xor(s1, off); s2 += __shfl_xor(s2, off); }
  const float mu = s1 * (1.f / 512.f);
  const float var = s2 * (1.f / 512.f) - mu * mu;
  const float rs = rsqrtf(var + 1e-5f);
#pragma unroll
  for (int i = 0; i < 4; ++i)
#pragma unroll
    for (int k = 0; k < 4; ++k) {
      int ch = (lane + i * 32) * 4 + k;
      Y[(size_t)token * 512 + ch] = (_Float16)((xv[4*i+k] - mu) * rs * wgt[ch] + bia[ch]);
    }
}

// ---------------------------------------------------------------------------
// WMMA GEMM, LDS-staged + double-buffered: out[M,N] = A[M,K](f16) * W[N,K]^T
// Block: 256 thr = 8 waves (4 M x 2 N), block tile 128x64, K-step 32.
// LDS: A slab 128x32 (reused by 2 N-waves), B slab 64x32 (reused by 4 M-waves).
// Global prefetch of slab k+1 is issued before the WMMAs on slab k.
// MODE 0: f16 store            MODE 1: fp32 += acc + bias (residual)
// MODE 2: f16 = gelu(acc+bias) MODE 3: same as 1
// ---------------------------------------------------------------------------
template <int MODE>
__global__ __launch_bounds__(256) void gemm_wmma(const _Float16* __restrict__ A,
                                                 const _Float16* __restrict__ Bw,
                                                 const float* __restrict__ bias,
                                                 _Float16* __restrict__ outH,
                                                 float* __restrict__ outF,
                                                 int M, int N, int K,
                                                 int lda, int ldb, int ldo) {
  __shared__ __align__(16) _Float16 As[2][128 * 32];  // 2 x 8 KB
  __shared__ __align__(16) _Float16 Bs[2][64 * 32];   // 2 x 4 KB

  const int tid  = threadIdx.x;
  const int lane = tid & 31;
  const int wv   = tid >> 5;       // 0..7
  const int hf   = lane >> 4;
  const int l15  = lane & 15;
  const int wm   = wv >> 1;        // 0..3  (M sub-tile)
  const int wn   = wv & 1;         // 0..1  (N sub-tile)
  const int mblk = blockIdx.y * 128;
  const int nblk = blockIdx.x * 64;

  // cooperative staging indices: A 128 rows x 32 cols, B 64 rows x 32 cols
  const int arow = tid >> 1;            // 0..127
  const int acol = (tid & 1) * 16;      // 0 / 16
  const int brow = tid >> 2;            // 0..63
  const int bcol = (tid & 3) * 8;       // 0/8/16/24

  const _Float16* gA = A + (size_t)(mblk + arow) * lda + acol;
  const _Float16* gB = Bw + (size_t)(nblk + brow) * ldb + bcol;

  const int nk = K >> 5;

  // prologue: stage k-slab 0
  {
    v8h a0 = *(const v8h*)gA;
    v8h a1 = *(const v8h*)(gA + 8);
    v8h b0 = *(const v8h*)gB;
    *(v8h*)(As[0] + arow * 32 + acol)     = a0;
    *(v8h*)(As[0] + arow * 32 + acol + 8) = a1;
    *(v8h*)(Bs[0] + brow * 32 + bcol)     = b0;
  }

  v8f acc[2][2] = {};
  int cur = 0;
  for (int kt = 0; kt < nk; ++kt) {
    __syncthreads();
    // global prefetch of next slab into registers (overlaps the WMMAs below)
    v8h na0, na1, nb0;
    if (kt + 1 < nk) {
      const _Float16* pA = gA + (kt + 1) * 32;
      const _Float16* pB = gB + (kt + 1) * 32;
      na0 = *(const v8h*)pA;
      na1 = *(const v8h*)(pA + 8);
      nb0 = *(const v8h*)pB;
      if (kt + 2 < nk) {
        __builtin_prefetch(gA + (kt + 2) * 32, 0, 3);
        __builtin_prefetch(gB + (kt + 2) * 32, 0, 3);
      }
    }
    // fragments from LDS[cur]
    v16h af[2], bf[2];
#pragma unroll
    for (int i = 0; i < 2; ++i) {
      const _Float16* pa = As[cur] + (wm * 32 + i * 16 + l15) * 32 + 8 * hf;
      af[i] = cat_v8h(*(const v8h*)pa, *(const v8h*)(pa + 16));
      const _Float16* pb = Bs[cur] + (wn * 32 + i * 16 + l15) * 32 + 8 * hf;
      bf[i] = cat_v8h(*(const v8h*)pb, *(const v8h*)(pb + 16));
    }
#pragma unroll
    for (int i = 0; i < 2; ++i)
#pragma unroll
      for (int j = 0; j < 2; ++j)
        acc[i][j] = __builtin_amdgcn_wmma_f32_16x16x32_f16(
            false, af[i], false, bf[j], (short)0, acc[i][j], false, false);
    // stage next slab into the other buffer (safe: its readers synced above)
    if (kt + 1 < nk) {
      const int nxt = cur ^ 1;
      *(v8h*)(As[nxt] + arow * 32 + acol)     = na0;
      *(v8h*)(As[nxt] + arow * 32 + acol + 8) = na1;
      *(v8h*)(Bs[nxt] + brow * 32 + bcol)     = nb0;
      cur = nxt;
    }
  }

  // epilogue
  const int m0 = mblk + wm * 32;
  const int n0 = nblk + wn * 32;
#pragma unroll
  for (int i = 0; i < 2; ++i)
#pragma unroll
    for (int j = 0; j < 2; ++j)
#pragma unroll
      for (int v = 0; v < 8; ++v) {
        const int m = m0 + i * 16 + 8 * hf + v;
        const int n = n0 + j * 16 + l15;
        float val = acc[i][j][v];
        if (MODE == 1 || MODE == 3) {
          outF[(size_t)m * ldo + n] += val + bias[n];
        } else if (MODE == 2) {
          float t = val + bias[n];
          t = 0.5f * t * (1.0f + erff(t * 0.70710678118654752f));
          outH[(size_t)m * ldo + n] = (_Float16)t;
        } else {
          outH[(size_t)m * ldo + n] = (_Float16)val;
        }
      }
}

// ---------------------------------------------------------------------------
// Cross-shaped window attention. One wave per (branch, window, head).
// qkv: f16 [T=32768, 3*512] (plane-major: q|k|v each 512 channels).
// att: f16 [T, 512] output (branch 0 -> cols 0..255, branch 1 -> 256..511).
// ---------------------------------------------------------------------------
__global__ __launch_bounds__(32) void attn_kernel(const _Float16* __restrict__ qkv,
                                                  _Float16* __restrict__ att) {
  __shared__ __align__(16) _Float16 qs[64 * 32];    //  4 KB  q window (scaled)
  __shared__ __align__(16) _Float16 ksT[192 * 32];  // 12 KB  k strips, [j][d]
  __shared__ __align__(16) _Float16 vs[32 * 192];   // 12 KB  v strips, [d][j]
  __shared__ __align__(16) float    simS[16 * 192]; // 12 KB  sim row-block
  __shared__ __align__(16) _Float16 aS[16 * 192];   //  6 KB  attn row-block

  const int lane = threadIdx.x;
  const int hf = lane >> 4;
  const int l15 = lane & 15;
  const int win = blockIdx.x;   // 0..511
  const int hd  = blockIdx.y;   // 0..7
  const int br  = blockIdx.z;   // 0: 'x' strips, 1: 'y' strips
  const int b   = win >> 6;
  const int wi  = (win >> 3) & 7;
  const int wj  = win & 7;
  const int chBase = br * 256 + hd * 32;
  const float scale = 0.17677669529663687f;  // (d=32)^-0.5

  // gather q (scaled)
  for (int idx = lane; idx < 64 * 32; idx += 32) {
    const int n = idx >> 5, dd = idx & 31;
    const int tok = b * 4096 + (wi * 8 + (n >> 3)) * 64 + wj * 8 + (n & 7);
    qs[idx] = (_Float16)((float)qkv[(size_t)tok * 1536 + chBase + dd] * scale);
  }
  // gather cross-shaped k/v strips (zero beyond image, as in reference)
  for (int idx = lane; idx < 192 * 32; idx += 32) {
    const int t = idx >> 5, dd = idx & 31;
    int y, x;
    if (br == 0) { const int wy = t / 24, xx = t % 24; y = wi * 8 + wy; x = wj * 8 + xx - 8; }
    else         { const int yy = t / 8,  wx = t % 8;  y = wi * 8 + yy - 8; x = wj * 8 + wx; }
    _Float16 kv = (_Float16)0.f, vv = (_Float16)0.f;
    if (y >= 0 && y < 64 && x >= 0 && x < 64) {
      const size_t tok = (size_t)(b * 4096 + y * 64 + x) * 1536;
      kv = qkv[tok + 512 + chBase + dd];
      vv = qkv[tok + 1024 + chBase + dd];
    }
    ksT[t * 32 + dd] = kv;
    vs[dd * 192 + t] = vv;
  }
  __syncthreads();

  for (int mt = 0; mt < 4; ++mt) {
    // A fragment: 16 q rows, K = d = 32
    const _Float16* pq = qs + (mt * 16 + l15) * 32 + 8 * hf;
    const v16h aq = cat_v8h(*(const v8h*)pq, *(const v8h*)(pq + 16));

    // stage 1: sim = q @ k  (12 N-tiles of 16)
    for (int nt = 0; nt < 12; ++nt) {
      const _Float16* pk = ksT + (nt * 16 + l15) * 32 + 8 * hf;
      const v16h bk = cat_v8h(*(const v8h*)pk, *(const v8h*)(pk + 16));
      v8f s = {};
      s = __builtin_amdgcn_wmma_f32_16x16x32_f16(false, aq, false, bk, (short)0, s, false, false);
#pragma unroll
      for (int v = 0; v < 8; ++v)
        simS[(8 * hf + v) * 192 + nt * 16 + l15] = s[v];
    }
    __syncthreads();

    // softmax over 192 per row; 2 lanes/row, combine via shfl_xor(16)
    {
      const int r = l15, base = hf * 96;
      float mx = -3.4e38f;
      for (int t = 0; t < 96; ++t) mx = fmaxf(mx, simS[r * 192 + base + t]);
      mx = fmaxf(mx, __shfl_xor(mx, 16));
      float sum = 0.f;
      for (int t = 0; t < 96; ++t) {
        const float e = __expf(simS[r * 192 + base + t] - mx);
        simS[r * 192 + base + t] = e;
        sum += e;
      }
      sum += __shfl_xor(sum, 16);
      const float inv = 1.f / sum;
      for (int t = 0; t < 96; ++t)
        aS[r * 192 + base + t] = (_Float16)(simS[r * 192 + base + t] * inv);
    }
    __syncthreads();

    // stage 2: out = attn @ v^T  (K = 192 in 6 chunks, 2 d-tiles)
    v8f o0 = {}, o1 = {};
    for (int kc = 0; kc < 6; ++kc) {
      const int k0 = kc * 32;
      const _Float16* pa = aS + l15 * 192 + k0 + 8 * hf;
      const v16h aa = cat_v8h(*(const v8h*)pa, *(const v8h*)(pa + 16));
      const _Float16* pv0 = vs + l15 * 192 + k0 + 8 * hf;
      const v16h b0 = cat_v8h(*(const v8h*)pv0, *(const v8h*)(pv0 + 16));
      const _Float16* pv1 = vs + (16 + l15) * 192 + k0 + 8 * hf;
      const v16h b1 = cat_v8h(*(const v8h*)pv1, *(const v8h*)(pv1 + 16));
      o0 = __builtin_amdgcn_wmma_f32_16x16x32_f16(false, aa, false, b0, (short)0, o0, false, false);
      o1 = __builtin_amdgcn_wmma_f32_16x16x32_f16(false, aa, false, b1, (short)0, o1, false, false);
    }

    // scatter to [T, 512] att matrix
#pragma unroll
    for (int v = 0; v < 8; ++v) {
      const int n = mt * 16 + 8 * hf + v;
      const int tok = b * 4096 + (wi * 8 + (n >> 3)) * 64 + wj * 8 + (n & 7);
      att[(size_t)tok * 512 + chBase + l15]      = (_Float16)o0[v];
      att[(size_t)tok * 512 + chBase + 16 + l15] = (_Float16)o1[v];
    }
    __syncthreads();
  }
}

// ---------------------------------------------------------------------------
// LePE: depthwise 3x3 conv on each 8x8 v-window (zero pad at window edge),
// accumulated into the att matrix. One thread per (br, window, ch, n).
// ---------------------------------------------------------------------------
__global__ __launch_bounds__(256) void lepe_kernel(const _Float16* __restrict__ qkv,
                                                   const float* __restrict__ cw,
                                                   const float* __restrict__ cb,
                                                   _Float16* __restrict__ att) {
  const int id = blockIdx.x * 256 + threadIdx.x;   // 2^24 total
  const int n  = id & 63;
  const int ch = (id >> 6) & 255;
  const int w  = (id >> 14) & 511;
  const int br = (id >> 23) & 1;
  const int b  = w >> 6;
  const int wi = (w >> 3) & 7;
  const int wj = w & 7;
  const int wy = n >> 3, wx = n & 7;

  float acc = cb[ch];
#pragma unroll
  for (int ky = 0; ky < 3; ++ky)
#pragma unroll
    for (int kx = 0; kx < 3; ++kx) {
      const int vy = wy + ky - 1, vx = wx + kx - 1;
      if (vy >= 0 && vy < 8 && vx >= 0 && vx < 8) {
        const int tok = b * 4096 + (wi * 8 + vy) * 64 + wj * 8 + vx;
        acc += (float)qkv[(size_t)tok * 1536 + 1024 + br * 256 + ch] * cw[ch * 9 + ky * 3 + kx];
      }
    }
  const int tok = b * 4096 + (wi * 8 + wy) * 64 + wj * 8 + wx;
  const size_t di = (size_t)tok * 512 + br * 256 + ch;
  att[di] = (_Float16)((float)att[di] + acc);
}

// ---------------------------------------------------------------------------
extern "C" void kernel_launch(void* const* d_in, const int* in_sizes, int n_in,
                              void* d_out, int out_size, void* d_ws, size_t ws_size,
                              hipStream_t stream) {
  (void)in_sizes; (void)n_in; (void)out_size; (void)ws_size;
  const float* x     = (const float*)d_in[0];
  const float* n1w   = (const float*)d_in[1];
  const float* n1b   = (const float*)d_in[2];
  const float* qkvw  = (const float*)d_in[3];
  const float* gvw   = (const float*)d_in[4];
  const float* gvb   = (const float*)d_in[5];
  const float* projw = (const float*)d_in[6];
  const float* projb = (const float*)d_in[7];
  const float* n2w   = (const float*)d_in[8];
  const float* n2b   = (const float*)d_in[9];
  const float* fc1w  = (const float*)d_in[10];
  const float* fc1b  = (const float*)d_in[11];
  const float* fc2w  = (const float*)d_in[12];
  const float* fc2b  = (const float*)d_in[13];
  float* out = (float*)d_out;

  // workspace layout (phase-overlaid): total ~241 MB
  char* ws = (char*)d_ws;
  float*    xt  = (float*)ws;                                            // 64 MiB residual stream
  _Float16* big = (_Float16*)(ws + (size_t)67108864);                    // 128 MiB: qkv_h then h1
  _Float16* act = (_Float16*)(ws + (size_t)67108864 + 134217728);        // 32 MiB: img / att / hdn
  _Float16* wqh = (_Float16*)(ws + (size_t)67108864 + 134217728 + 33554432);
  _Float16* wph = wqh + 1536 * 512;
  _Float16* w1h = wph + 512 * 512;
  _Float16* w2h = w1h + 2048 * 512;

  // 1. x [B,C,HW] -> xt [B,HW,C]
  transpose_kernel<<<dim3(4096 / 32, 512 / 32, 8), dim3(32, 8), 0, stream>>>(x, xt, 512, 4096);
  // 2. weights to f16
  f32_to_f16<<<(1536 * 512 + 255) / 256, 256, 0, stream>>>(qkvw, wqh, 1536 * 512);
  f32_to_f16<<<(512 * 512 + 255) / 256, 256, 0, stream>>>(projw, wph, 512 * 512);
  f32_to_f16<<<(2048 * 512 + 255) / 256, 256, 0, stream>>>(fc1w, w1h, 2048 * 512);
  f32_to_f16<<<(512 * 2048 + 255) / 256, 256, 0, stream>>>(fc2w, w2h, 512 * 2048);
  // 3. LN1 -> act (img, f16)
  layernorm_kernel<<<4096, 256, 0, stream>>>(xt, n1w, n1b, act);
  // 4. qkv = img @ qkv_w^T  -> big [T,1536] f16
  gemm_wmma<0><<<dim3(1536 / 64, 32768 / 128), 256, 0, stream>>>(
      act, wqh, nullptr, big, nullptr, 32768, 1536, 512, 512, 512, 1536);
  // 5. cross-shaped window attention -> act (att, f16)
  attn_kernel<<<dim3(512, 8, 2), 32, 0, stream>>>(big, act);
  // 6. LePE conv accumulated into att
  lepe_kernel<<<65536, 256, 0, stream>>>(big, gvw, gvb, act);
  // 7. xt += att @ proj_w^T + proj_b
  gemm_wmma<1><<<dim3(512 / 64, 32768 / 128), 256, 0, stream>>>(
      act, wph, projb, nullptr, xt, 32768, 512, 512, 512, 512, 512);
  // 8. LN2 -> act (hdn, f16)
  layernorm_kernel<<<4096, 256, 0, stream>>>(xt, n2w, n2b, act);
  // 9. h1 = gelu(hdn @ fc1_w^T + fc1_b) -> big [T,2048] f16
  gemm_wmma<2><<<dim3(2048 / 64, 32768 / 128), 256, 0, stream>>>(
      act, w1h, fc1b, big, nullptr, 32768, 2048, 512, 512, 512, 2048);
  // 10. xt += h1 @ fc2_w^T + fc2_b
  gemm_wmma<3><<<dim3(512 / 64, 32768 / 128), 256, 0, stream>>>(
      big, w2h, fc2b, nullptr, xt, 32768, 512, 2048, 2048, 2048, 512);
  // 11. xt [B,HW,C] -> out [B,C,H,W]
  transpose_kernel<<<dim3(512 / 32, 4096 / 32, 8), dim3(32, 8), 0, stream>>>(xt, out, 4096, 512);
}
